// TransformerBlock_85581518340338
// MI455X (gfx1250) — matrix operand order repaired and checked by reading, not compiled, not run on hardware
//
#include <hip/hip_runtime.h>
#include <math.h>

typedef __attribute__((ext_vector_type(16))) __bf16 v16bf;
typedef __attribute__((ext_vector_type(8)))  float  v8f;
typedef __attribute__((ext_vector_type(4)))  int    int4v;

#define NPOS 65536L

#if __has_builtin(__builtin_amdgcn_global_load_async_to_lds_b128)
#define HAVE_ASYNC_LDS 1
#else
#define HAVE_ASYNC_LDS 0
#endif

typedef __attribute__((address_space(1))) int4v* as1_i4p;   // global int4
typedef __attribute__((address_space(3))) int4v* as3_i4p;   // LDS int4

#if __has_builtin(__builtin_amdgcn_s_wait_asynccnt)
#define WAIT_ASYNC(n) __builtin_amdgcn_s_wait_asynccnt(n)
#else
#define WAIT_ASYNC(n) asm volatile("s_wait_asynccnt %0" ::"i"(n) : "memory")
#endif

union FragBF { v16bf v; uint4 u[2]; };

__device__ __forceinline__ v8f wmma_bf16(v16bf a, v16bf b, v8f c) {
  return __builtin_amdgcn_wmma_f32_16x16x32_bf16(false, a, false, b, (short)0, c,
                                                 false, false);
}

// ---------------------------------------------------------------------------
// Pack fp32 matrix [M x K] -> bf16 [M x Kpad], zero-padding k >= K
// ---------------------------------------------------------------------------
__global__ __launch_bounds__(256) void k_pack_bf16(
    const float* __restrict__ A, __bf16* __restrict__ out,
    int K, int Kpad, long total)
{
  const long i = (long)blockIdx.x * 256 + threadIdx.x;
  if (i >= total) return;
  const long m = i / Kpad;
  const int  k = (int)(i - m * Kpad);
  out[i] = (k < K) ? (__bf16)A[m * K + k] : (__bf16)0.f;
}

// ---------------------------------------------------------------------------
// LayerNorm over 48 channels per position -> bf16 activations
// ---------------------------------------------------------------------------
__global__ __launch_bounds__(256) void k_layernorm(
    const float* __restrict__ x, const float* __restrict__ g,
    const float* __restrict__ bta, __bf16* __restrict__ y)
{
  const long p = (long)blockIdx.x * 256 + threadIdx.x;
  const float* xb = x + (long)blockIdx.y * 48 * NPOS;
  __bf16* yb = y + (long)blockIdx.y * 48 * NPOS;
  float v[48];
  float mu = 0.f;
  #pragma unroll
  for (int c = 0; c < 48; ++c) { v[c] = xb[(long)c * NPOS + p]; mu += v[c]; }
  mu *= (1.f / 48.f);
  float var = 0.f;
  #pragma unroll
  for (int c = 0; c < 48; ++c) { float d = v[c] - mu; var += d * d; }
  const float inv = rsqrtf(var * (1.f / 48.f) + 1e-5f);
  #pragma unroll
  for (int c = 0; c < 48; ++c)
    yb[(long)c * NPOS + p] = (__bf16)((v[c] - mu) * inv * g[c] + bta[c]);
}

// ---------------------------------------------------------------------------
// Generic 1x1-conv GEMM:  Y[m,n] = sum_k A[m,k] * X[k,n]  (+bias, +residual)
// A: packed bf16 [M x Kpad] (zero-padded); X: bf16, row stride NPOS.
// Block: NTW 16-column tiles; X staged transposed in LDS (zero-padded K).
// WPN = 8/NTW waves cooperate per n-tile; wave owns m-tiles ms, ms+WPN, ...
// mode 0: bf16 store; mode 1: fp32 store with residual add.
// ---------------------------------------------------------------------------
template <int NTW>
__global__ __launch_bounds__(256) void k_gemm_wmma(
    const __bf16* __restrict__ Abf, const float* __restrict__ bias,
    const __bf16* __restrict__ X, const float* __restrict__ R,
    void* __restrict__ O,
    int M, int K, int ydiv,
    long aS1, long aS2, long xS1, long xS2, long oS1, long oS2, long rS1,
    int mode)
{
  extern __shared__ char smem[];
  __bf16* ldsT = (__bf16*)smem;                 // [16*NTW cols][Kpad]
  const int Kpad = (K + 31) & ~31;
  const long n0 = (long)blockIdx.x * (16 * NTW);
  const int yb = blockIdx.y / ydiv, yh = blockIdx.y % ydiv;
  Abf += yb * aS1 + yh * aS2;
  X   += yb * xS1 + yh * xS2;
  const long oOff = yb * oS1 + yh * oS2;
  const long rOff = yb * rS1;
  const int tid = threadIdx.x;

  // cooperative transposed load X[k][n0+jj] -> ldsT[jj][k], zero-fill k >= K
  constexpr int CW = 4 * NTW;                   // 4-element chunks per k row
  for (int c = tid; c < Kpad * CW; c += 256) {
    const int k = c / CW, jj = (c % CW) * 4;
    __bf16 t4[4];
    if (k < K) {
      *(uint2*)t4 = *(const uint2*)&X[(long)k * NPOS + n0 + jj];
    } else {
      uint2 z; z.x = 0u; z.y = 0u; *(uint2*)t4 = z;
    }
    #pragma unroll
    for (int i = 0; i < 4; ++i) ldsT[(jj + i) * Kpad + k] = t4[i];
  }
  __syncthreads();

  const int wave = tid >> 5, lane = tid & 31;
  const int hf = lane >> 4, mrow = lane & 15;
  constexpr int WPN = 8 / NTW;                  // waves per n-tile
  const int nt = wave / WPN;
  const int ms = wave % WPN;
  const int nMt = M >> 4;

  const v8f vz = {0.f, 0.f, 0.f, 0.f, 0.f, 0.f, 0.f, 0.f};
  v8f acc[5];
  int mts[5], nm = 0;
  for (int mt = ms; mt < nMt && nm < 5; mt += WPN) {
    mts[nm] = mt; acc[nm] = vz; ++nm;
  }

  const __bf16* brow = &ldsT[(nt * 16 + mrow) * Kpad];
  for (int kk = 0; kk < Kpad; kk += 32) {
    FragBF b;                                   // B[k,n]: k = kk + e + 16*hf, n = col
    b.u[0] = *(const uint4*)(brow + kk + 16 * hf);
    b.u[1] = *(const uint4*)(brow + kk + 16 * hf + 8);
    #pragma unroll
    for (int i = 0; i < 5; ++i) {
      if (i >= nm) break;
      const __bf16* Arow = Abf + (long)(mts[i] * 16 + mrow) * Kpad;
      FragBF a;                                 // A[m,k]: k = kk + e + 16*hf
      a.u[0] = *(const uint4*)(Arow + kk + 16 * hf);
      a.u[1] = *(const uint4*)(Arow + kk + 16 * hf + 8);
      acc[i] = wmma_bf16(a.v, b.v, acc[i]);
    }
  }

  #pragma unroll
  for (int i = 0; i < 5; ++i) {
    if (i >= nm) break;
    const int m0 = mts[i] * 16;
    #pragma unroll
    for (int r = 0; r < 8; ++r) {
      const int ch = m0 + r + 8 * hf;           // D layout: m = r + 8*half
      const long col = n0 + nt * 16 + mrow;
      float val = acc[i][r] + (bias ? bias[ch] : 0.f);
      if (mode == 0) {
        ((__bf16*)O)[oOff + (long)ch * NPOS + col] = (__bf16)val;
      } else {
        ((float*)O)[oOff + (long)ch * NPOS + col] =
            val + R[rOff + (long)ch * NPOS + col];
      }
    }
  }
}

// ---------------------------------------------------------------------------
// Grouped depthwise 3x3x3 conv for qkv: 576 ch, groups of 4-in/4-out
// ---------------------------------------------------------------------------
__global__ __launch_bounds__(256) void k_dw_qkv(
    const __bf16* __restrict__ in, const float* __restrict__ w,
    const float* __restrict__ bias, __bf16* __restrict__ out)
{
  const long p = (long)blockIdx.x * 256 + threadIdx.x;
  const int o = blockIdx.y, b = blockIdx.z;
  const int t = (int)(p >> 12), hy = (int)((p >> 6) & 63), wx = (int)(p & 63);
  const int gbase = (o >> 2) << 2;
  const float* wp = w + (long)o * 108;
  float acc = bias[o];
  for (int i = 0; i < 4; ++i) {
    const __bf16* ip = in + ((long)b * 576 + gbase + i) * NPOS;
    #pragma unroll
    for (int dt = -1; dt <= 1; ++dt) {
      const int tt = t + dt; if (tt < 0 || tt > 15) continue;
      #pragma unroll
      for (int dh = -1; dh <= 1; ++dh) {
        const int hh = hy + dh; if (hh < 0 || hh > 63) continue;
        #pragma unroll
        for (int dw2 = -1; dw2 <= 1; ++dw2) {
          const int ww = wx + dw2; if (ww < 0 || ww > 63) continue;
          acc += wp[((i * 3 + dt + 1) * 3 + dh + 1) * 3 + dw2 + 1] *
                 (float)ip[((long)tt << 12) + (hh << 6) + ww];
        }
      }
    }
  }
  out[((long)b * 576 + o) * NPOS + p] = (__bf16)acc;
}

// ---------------------------------------------------------------------------
// Per-channel L2 norm of q,k (channels 0..383 of qkvdw) -> 1/max(norm,1e-12)
// ---------------------------------------------------------------------------
__global__ __launch_bounds__(256) void k_colnorm(
    const __bf16* __restrict__ qkv, float* __restrict__ scales)
{
  const int b = blockIdx.x / 384, ch = blockIdx.x % 384;
  const __bf16* p = qkv + ((long)b * 576 + ch) * NPOS;
  float s = 0.f;
  for (long i = threadIdx.x; i < NPOS; i += 256) {
    const float v = (float)p[i]; s += v * v;
  }
  __shared__ float sm[256];
  sm[threadIdx.x] = s; __syncthreads();
  for (int st = 128; st > 0; st >>= 1) {
    if (threadIdx.x < st) sm[threadIdx.x] += sm[threadIdx.x + st];
    __syncthreads();
  }
  if (threadIdx.x == 0) scales[blockIdx.x] = 1.f / fmaxf(sqrtf(sm[0]), 1e-12f);
}

// ---------------------------------------------------------------------------
// attn_raw[b,h,c,d] = (sum_n q[c,n] k[d,n]) * s_q[c] * s_k[d] * temp[h]
// grid: x = 3x3 (c,d) tiles, y = b*4+h. 8 waves split K; LDS cross-wave reduce.
// K chunks staged with async global->LDS loads, double-buffered (ASYNCcnt).
// ---------------------------------------------------------------------------
__global__ __launch_bounds__(256) void k_attn_qk(
    const __bf16* __restrict__ qkv, const float* __restrict__ scales,
    const float* __restrict__ temp, float* __restrict__ attn)
{
  const int b = blockIdx.y >> 2, h = blockIdx.y & 3;
  const int ct = blockIdx.x / 3, dt = blockIdx.x % 3;
  const __bf16* q  = qkv + ((long)b * 576 +       h * 48 + ct * 16) * NPOS;
  const __bf16* kp = qkv + ((long)b * 576 + 192 + h * 48 + dt * 16) * NPOS;

  __shared__ __bf16 qs[2][16 * 256];
  __shared__ __bf16 ks[2][16 * 256];
  __shared__ float red[8][256];

  const int tid = threadIdx.x, wave = tid >> 5, lane = tid & 31;
  const int hf = lane >> 4, mrow = lane & 15;
  v8f acc = {0.f, 0.f, 0.f, 0.f, 0.f, 0.f, 0.f, 0.f};
  const int kw = wave * 32;
  const int iters = (int)(NPOS / 256);

#if HAVE_ASYNC_LDS
  // 4 async b128 ops per wave per chunk (2 chunk slots x 2 tensors)
  auto issue = [&](long kk0, int buf) {
    #pragma unroll
    for (int c = tid; c < 512; c += 256) {
      const int row = c >> 5, col = (c & 31) << 3;
      __builtin_amdgcn_global_load_async_to_lds_b128(
          (as1_i4p)&q[(long)row * NPOS + kk0 + col],
          (as3_i4p)&qs[buf][row * 256 + col], 0, 0);
      __builtin_amdgcn_global_load_async_to_lds_b128(
          (as1_i4p)&kp[(long)row * NPOS + kk0 + col],
          (as3_i4p)&ks[buf][row * 256 + col], 0, 0);
    }
  };
  issue(0, 0);
  for (int it = 0; it < iters; ++it) {
    const int buf = it & 1;
    if (it + 1 < iters) {
      issue((long)(it + 1) * 256, buf ^ 1);
      WAIT_ASYNC(4);                            // current chunk landed in LDS
    } else {
      WAIT_ASYNC(0);
    }
    __syncthreads();
    FragBF a, bb;
    const __bf16* qrow = &qs[buf][mrow * 256];  // row = c, contiguous K
    const __bf16* krow = &ks[buf][mrow * 256];  // row = d, contiguous K
    a.u[0]  = *(const uint4*)(qrow + kw + 16 * hf);       // k = kw + e + 16*hf
    a.u[1]  = *(const uint4*)(qrow + kw + 16 * hf + 8);
    bb.u[0] = *(const uint4*)(krow + kw + 16 * hf);       // k = kw + e + 16*hf
    bb.u[1] = *(const uint4*)(krow + kw + 16 * hf + 8);
    acc = wmma_bf16(a.v, bb.v, acc);
    __syncthreads();                            // buf reusable for it+2's issue
  }
#else
  for (int it = 0; it < iters; ++it) {
    const long kk0 = (long)it * 256;
    for (int c = tid; c < 512; c += 256) {
      const int row = c >> 5, col = (c & 31) << 3;
      *(uint4*)&qs[0][row * 256 + col] = *(const uint4*)&q [(long)row * NPOS + kk0 + col];
      *(uint4*)&ks[0][row * 256 + col] = *(const uint4*)&kp[(long)row * NPOS + kk0 + col];
    }
    if (kk0 + 256 < NPOS) {                     // global_prefetch_b8 next chunk
      __builtin_prefetch(&q [(long)(tid & 15) * NPOS + kk0 + 256 + ((tid >> 4) << 4)], 0, 3);
      __builtin_prefetch(&kp[(long)(tid & 15) * NPOS + kk0 + 256 + ((tid >> 4) << 4)], 0, 3);
    }
    __syncthreads();
    FragBF a, bb;
    const __bf16* qrow = &qs[0][mrow * 256];
    const __bf16* krow = &ks[0][mrow * 256];
    a.u[0]  = *(const uint4*)(qrow + kw + 16 * hf);
    a.u[1]  = *(const uint4*)(qrow + kw + 16 * hf + 8);
    bb.u[0] = *(const uint4*)(krow + kw + 16 * hf);
    bb.u[1] = *(const uint4*)(krow + kw + 16 * hf + 8);
    acc = wmma_bf16(a.v, bb.v, acc);
    __syncthreads();
  }
#endif

  #pragma unroll
  for (int r = 0; r < 8; ++r) red[wave][r * 32 + lane] = acc[r];
  __syncthreads();
  if (wave == 0) {
    #pragma unroll
    for (int r = 0; r < 8; ++r) {
      float s = 0.f;
      #pragma unroll
      for (int w2 = 0; w2 < 8; ++w2) s += red[w2][r * 32 + lane];
      const int c = ct * 16 + r + 8 * hf;
      const int d = dt * 16 + mrow;
      s *= scales[b * 384 + h * 48 + c] * scales[b * 384 + 192 + h * 48 + d] * temp[h];
      attn[(((long)b * 4 + h) * 48 + c) * 48 + d] = s;
    }
  }
}

// ---------------------------------------------------------------------------
// Row softmax over 48 entries
// ---------------------------------------------------------------------------
__global__ __launch_bounds__(64) void k_softmax48(float* __restrict__ a)
{
  __shared__ float sm[64];
  const int t = threadIdx.x;
  float* p = a + (long)blockIdx.x * 48;
  const float v = (t < 48) ? p[t] : -3.0e38f;
  sm[t] = v; __syncthreads();
  for (int s = 32; s > 0; s >>= 1) { if (t < s) sm[t] = fmaxf(sm[t], sm[t + s]); __syncthreads(); }
  const float mx = sm[0]; __syncthreads();
  const float e = (t < 48) ? __expf(v - mx) : 0.f;
  sm[t] = e; __syncthreads();
  for (int s = 32; s > 0; s >>= 1) { if (t < s) sm[t] += sm[t + s]; __syncthreads(); }
  if (t < 48) p[t] = e / sm[0];
}

// ---------------------------------------------------------------------------
// FFN depthwise 3x3x3 (192 ch) fused with GELU gate: out = gelu(z1)*z2, 96 ch
// ---------------------------------------------------------------------------
__global__ __launch_bounds__(256) void k_dw_ffn_gate(
    const __bf16* __restrict__ z, const float* __restrict__ w,
    const float* __restrict__ bias, __bf16* __restrict__ gated)
{
  const long p = (long)blockIdx.x * 256 + threadIdx.x;
  const int c = blockIdx.y, b = blockIdx.z;
  const int t = (int)(p >> 12), hy = (int)((p >> 6) & 63), wx = (int)(p & 63);
  const __bf16* z1 = z + ((long)b * 192 + c) * NPOS;
  const __bf16* z2 = z + ((long)b * 192 + c + 96) * NPOS;
  const float* w1 = w + (long)c * 27;
  const float* w2 = w + (long)(c + 96) * 27;
  float a1 = bias[c], a2 = bias[c + 96];
  #pragma unroll
  for (int dt = -1; dt <= 1; ++dt) {
    const int tt = t + dt; if (tt < 0 || tt > 15) continue;
    #pragma unroll
    for (int dh = -1; dh <= 1; ++dh) {
      const int hh = hy + dh; if (hh < 0 || hh > 63) continue;
      #pragma unroll
      for (int dw2 = -1; dw2 <= 1; ++dw2) {
        const int ww = wx + dw2; if (ww < 0 || ww > 63) continue;
        const long idx = ((long)tt << 12) + (hh << 6) + ww;
        const int widx = ((dt + 1) * 3 + dh + 1) * 3 + dw2 + 1;
        a1 += w1[widx] * (float)z1[idx];
        a2 += w2[widx] * (float)z2[idx];
      }
    }
  }
  const float g = 0.5f * a1 * (1.f + erff(a1 * 0.70710678f));
  gated[((long)b * 96 + c) * NPOS + p] = (__bf16)(g * a2);
}

// ---------------------------------------------------------------------------
extern "C" void kernel_launch(void* const* d_in, const int* in_sizes, int n_in,
                              void* d_out, int out_size, void* d_ws, size_t ws_size,
                              hipStream_t stream)
{
  (void)in_sizes; (void)n_in; (void)out_size; (void)ws_size;
  const float* x       = (const float*)d_in[0];
  const float* ln1_w   = (const float*)d_in[1];
  const float* ln1_b   = (const float*)d_in[2];
  const float* qkv_w   = (const float*)d_in[3];
  const float* qkv_b   = (const float*)d_in[4];
  const float* qkvdw_w = (const float*)d_in[5];
  const float* qkvdw_b = (const float*)d_in[6];
  const float* temp    = (const float*)d_in[7];
  const float* proj_w  = (const float*)d_in[8];
  const float* proj_b  = (const float*)d_in[9];
  const float* ln2_w   = (const float*)d_in[10];
  const float* ln2_b   = (const float*)d_in[11];
  const float* pin_w   = (const float*)d_in[12];
  const float* pin_b   = (const float*)d_in[13];
  const float* dw_w    = (const float*)d_in[14];
  const float* dw_b    = (const float*)d_in[15];
  const float* pout_w  = (const float*)d_in[16];
  const float* pout_b  = (const float*)d_in[17];

  const long N = NPOS;
  char* ws = (char*)d_ws;
  size_t off = 0;
  auto alloc = [&](size_t bytes) -> char* {
    char* p = ws + off; off += (bytes + 255) & ~(size_t)255; return p;
  };
  __bf16* y_bf   = (__bf16*)alloc((size_t)2 * 48  * N * sizeof(__bf16));
  __bf16* qkvbuf = (__bf16*)alloc((size_t)2 * 576 * N * sizeof(__bf16)); // reused: attn_out, zbuf
  __bf16* qkvdw  = (__bf16*)alloc((size_t)2 * 576 * N * sizeof(__bf16)); // reused: gated
  float*  x1     = (float*) alloc((size_t)2 * 48  * N * sizeof(float));
  float*  attnm  = (float*) alloc((size_t)2 * 4 * 48 * 48 * sizeof(float));
  float*  scales = (float*) alloc((size_t)2 * 384 * sizeof(float));
  // packed bf16 A matrices (zero-padded Kpad)
  __bf16* qkvw_p  = (__bf16*)alloc((size_t)576 * 64  * sizeof(__bf16));
  __bf16* projw_p = (__bf16*)alloc((size_t)48  * 192 * sizeof(__bf16));
  __bf16* pinw_p  = (__bf16*)alloc((size_t)192 * 64  * sizeof(__bf16));
  __bf16* poutw_p = (__bf16*)alloc((size_t)48  * 96  * sizeof(__bf16));
  __bf16* attnm_p = (__bf16*)alloc((size_t)384 * 64  * sizeof(__bf16));
  __bf16* attn_out = qkvbuf;   // live after raw qkv consumed
  __bf16* zbuf     = qkvbuf;   // live after attn_out consumed
  __bf16* gated    = qkvdw;    // live after v consumed

  const dim3 blk(256);
  const int NB = (int)(N / 256);    // 256 position blocks

  // 0) pack weights to bf16 (Kpad layout)
  k_pack_bf16<<<dim3(144), blk, 0, stream>>>(qkv_w,  qkvw_p,  48,  64,  (long)576 * 64);
  k_pack_bf16<<<dim3(36),  blk, 0, stream>>>(proj_w, projw_p, 192, 192, (long)48 * 192);
  k_pack_bf16<<<dim3(48),  blk, 0, stream>>>(pin_w,  pinw_p,  48,  64,  (long)192 * 64);
  k_pack_bf16<<<dim3(18),  blk, 0, stream>>>(pout_w, poutw_p, 96,  96,  (long)48 * 96);

  // 1) LN1 -> bf16
  k_layernorm<<<dim3(NB, 2), blk, 0, stream>>>(x, ln1_w, ln1_b, y_bf);
  // 2) qkv 1x1 GEMM (M=576, K=48, NTW=1) -> qkvbuf bf16
  k_gemm_wmma<1><<<dim3((int)(N / 16), 2), blk, 64 * 16 * 2, stream>>>(
      qkvw_p, qkv_b, y_bf, nullptr, qkvbuf,
      576, 48, 1, 0, 0, 48 * N, 0, 576 * N, 0, 0, 0);
  // 3) grouped depthwise 3x3x3 -> qkvdw bf16
  k_dw_qkv<<<dim3(NB, 576, 2), blk, 0, stream>>>(qkvbuf, qkvdw_w, qkvdw_b, qkvdw);
  // 4) q,k channel norms
  k_colnorm<<<dim3(2 * 384), blk, 0, stream>>>(qkvdw, scales);
  // 5) attn = (q kT) * scales * temperature  (async double-buffered)
  k_attn_qk<<<dim3(9, 8), blk, 0, stream>>>(qkvdw, scales, temp, attnm);
  // 6) softmax over d, then pack attn to bf16 [384 x 64]
  k_softmax48<<<dim3(2 * 4 * 48), dim3(64), 0, stream>>>(attnm);
  k_pack_bf16<<<dim3(96), blk, 0, stream>>>(attnm, attnm_p, 48, 64, (long)384 * 64);
  // 7) out = attn @ v (per b,h: M=48, K=48, NTW=8) -> attn_out bf16
  k_gemm_wmma<8><<<dim3((int)(N / 128), 8), blk, 64 * 128 * 2, stream>>>(
      attnm_p, nullptr, qkvdw + (long)384 * N, nullptr, attn_out,
      48, 48, 4, 4 * 48 * 64, 48 * 64, 576 * N, 48 * N, 192 * N, 48 * N, 0, 0);
  // 8) proj (M=48, K=192, NTW=8) + residual x -> x1 fp32
  k_gemm_wmma<8><<<dim3((int)(N / 128), 2), blk, 192 * 128 * 2, stream>>>(
      projw_p, proj_b, attn_out, x, x1,
      48, 192, 1, 0, 0, 192 * N, 0, 48 * N, 0, 48 * N, 1);
  // 9) LN2 -> bf16
  k_layernorm<<<dim3(NB, 2), blk, 0, stream>>>(x1, ln2_w, ln2_b, y_bf);
  // 10) pin (M=192, K=48, NTW=2) -> zbuf bf16
  k_gemm_wmma<2><<<dim3((int)(N / 32), 2), blk, 64 * 32 * 2, stream>>>(
      pinw_p, pin_b, y_bf, nullptr, zbuf,
      192, 48, 1, 0, 0, 48 * N, 0, 192 * N, 0, 0, 0);
  // 11) FFN depthwise + GELU gate -> gated bf16 (96 ch)
  k_dw_ffn_gate<<<dim3(NB, 96, 2), blk, 0, stream>>>(zbuf, dw_w, dw_b, gated);
  // 12) pout (M=48, K=96, NTW=8) + residual x1 -> d_out fp32
  k_gemm_wmma<8><<<dim3((int)(N / 128), 2), blk, 96 * 128 * 2, stream>>>(
      poutw_p, pout_b, gated, x1, (float*)d_out,
      48, 96, 1, 0, 0, 96 * N, 0, 48 * N, 0, 48 * N, 1);
}
